// HGCLayer_53102975647844
// MI455X (gfx1250) — matrix operand order repaired and checked by reading, not compiled, not run on hardware
//
#include <hip/hip_runtime.h>
#include <math.h>
#include <stdint.h>

// ---------------------------------------------------------------------------
// HGC layer for MI455X (gfx1250, wave32).
// Edge MLP GEMM (E=320k) x (259->128) ~21 GFLOP -> v_wmma_f32_16x16x32_f16.
// Node tables are L2-resident (5 MB vs 192 MB L2); att_w1 is pre-swizzled to
// WMMA fragment-major order and held in LDS (73 KB of 320 KB/WGP), loaded
// once per persistent block with global_load_async_to_lds_b128. Fragment
// loads are single v16h (2x ds_load_b128/global_load_b128) per operand.
// Scatter uses global_atomic_add_f32 into an L2-resident accumulator.
// ---------------------------------------------------------------------------

typedef __attribute__((ext_vector_type(16))) _Float16 v16h;
typedef __attribute__((ext_vector_type(8)))  _Float16 v8h;
typedef __attribute__((ext_vector_type(8)))  float    v8f;

#define EPSV  1e-6f
#define NORMF 1000.0f
#define DD    128
#define KT    288      // att_in K (259) padded to 9*32 for f16 WMMA

// ---------------- helpers ----------------

__device__ __forceinline__ float blk_sum128(float v, float* red) {
  const int t = threadIdx.x;
  red[t] = v; __syncthreads();
  #pragma unroll
  for (int s = 64; s > 0; s >>= 1) {
    if (t < s) red[t] += red[t + s];
    __syncthreads();
  }
  float r = red[0]; __syncthreads();
  return r;
}

__device__ __forceinline__ float wave_sum32(float v) {
  #pragma unroll
  for (int m = 16; m > 0; m >>= 1) v += __shfl_xor(v, m, 32);
  return v;
}

// Pack index: fragment-major B layout.
// p = (((kk*8 + j)*32 + l)*16 + t) holds B[kk*32 + 8*(l>>4) + (t&7) + 16*(t>>3)][16*j + (l&15)]
// so lane l of wave j reads 32 contiguous bytes per K-block kk.

// ---------------- kernel 0: weight conversion/swizzle + accumulator zero ----------------

__global__ void k_init(const float* __restrict__ W, const float* __restrict__ w1,
                       _Float16* __restrict__ W16p, _Float16* __restrict__ w1pp,
                       float* __restrict__ agg, int total) {
  const int t = blockIdx.x * blockDim.x + threadIdx.x;
  if (t < DD * DD) {          // W (128x128): 4 K-blocks
    const int tt = t & 15, l = (t >> 4) & 31, j = (t >> 9) & 7, kk = t >> 12;
    const int k = kk * 32 + 8 * (l >> 4) + (tt & 7) + 16 * (tt >> 3);
    const int n = 16 * j + (l & 15);
    W16p[t] = (_Float16)W[k * DD + n];
  }
  if (t < KT * DD) {          // att_w1 padded (288x128): 9 K-blocks
    const int tt = t & 15, l = (t >> 4) & 31, j = (t >> 9) & 7, kk = t >> 12;
    const int k = kk * 32 + 8 * (l >> 4) + (tt & 7) + 16 * (tt >> 3);
    const int n = 16 * j + (l & 15);
    w1pp[t] = (k < 259) ? (_Float16)w1[k * DD + n] : (_Float16)0.0f;
  }
  if (t < total) agg[t] = 0.0f;
}

// ---------------- kernel 1: u = logmap0(h)  (elementwise, f32->f16) ----------------

__global__ void k_node_pre(const float* __restrict__ h, _Float16* __restrict__ u16,
                           int total) {
  const int t = blockIdx.x * blockDim.x + threadIdx.x;
  if (t >= total) return;
  const int i = t >> 7, d = t & 127;
  const float x0 = fmaxf(h[(size_t)i * DD], 1.0f + EPSV);
  const float coef = acoshf(x0) * rsqrtf(x0 * x0 - 1.0f);
  u16[t] = (_Float16)((d == 0) ? 0.0f : coef * h[t]);
}

// ---------------- kernel 2: xt = u @ W via WMMA (16-node tiles) ----------------

__global__ void __launch_bounds__(256) k_gemm_xw(const _Float16* __restrict__ u16,
                                                 const _Float16* __restrict__ W16p,
                                                 float* __restrict__ xt, int N) {
  __shared__ _Float16 A[16 * DD];
  const int tid = threadIdx.x;
  const int node0 = blockIdx.x * 16;
  for (int q = tid; q < 16 * DD; q += 256) {
    int nid = node0 + (q >> 7);
    if (nid >= N) nid = N - 1;
    A[q] = u16[(size_t)nid * DD + (q & 127)];
  }
  __syncthreads();
  const int w = tid >> 5, l = tid & 31, hi = l >> 4, nn = l & 15, nb = 16 * w;
  v8f c = {};
  #pragma unroll
  for (int kk = 0; kk < 4; ++kk) {
    const int k0 = kk * 32;
    v16h a;
    const _Float16* Ar = A + (l & 15) * DD + k0;
    const v8h a0 = *(const v8h*)(Ar + 8 * hi);
    const v8h a1 = *(const v8h*)(Ar + 16 + 8 * hi);
    #pragma unroll
    for (int j = 0; j < 8; ++j) { a[j] = a0[j]; a[8 + j] = a1[j]; }
    const v16h b = *(const v16h*)(W16p + (size_t)(((kk * 8 + w) * 32 + l) * 16));
    c = __builtin_amdgcn_wmma_f32_16x16x32_f16(false, a, false, b, (short)0, c, false, false);
  }
  if (node0 + 16 <= N) {
    #pragma unroll
    for (int r = 0; r < 8; ++r)
      xt[(size_t)(node0 + r + 8 * hi) * DD + nb + nn] = c[r];
  } else {
    #pragma unroll
    for (int r = 0; r < 8; ++r) {
      const int nid = node0 + r + 8 * hi;
      if (nid < N) xt[(size_t)nid * DD + nb + nn] = c[r];
    }
  }
}

// ---------------- kernel 3: x = expmap(expmap0(xt), transp0(bias)); xtan = logmap0(x) ----------------

__global__ void __launch_bounds__(128) k_node_mid(const float* __restrict__ xt,
                                                  const float* __restrict__ bias,
                                                  float* __restrict__ x,
                                                  _Float16* __restrict__ xtan16) {
  __shared__ float red[128];
  const int i = blockIdx.x, d = threadIdx.x;
  const float v = xt[(size_t)i * DD + d];
  // expmap0(proj_tan0(xt))
  const float sp2 = blk_sum128((d == 0) ? 0.0f : v * v, red);
  const float n1 = sqrtf(fmaxf(sp2, EPSV));
  const float x0 = coshf(n1);
  const float xd = (d == 0) ? x0 : (sinhf(n1) / n1) * v;
  // b = transp0(x, proj_tan0(bias))
  const float bsp = bias[d];
  const float bip = blk_sum128((d == 0) ? 0.0f : xd * bsp, red);
  const float coefb = bip / (1.0f + x0);
  const float bd = (d == 0) ? coefb * (x0 + 1.0f) : (bsp + coefb * xd);
  // x = expmap(x, b)
  const float bb = blk_sum128((d == 0) ? -bd * bd : bd * bd, red);
  const float n2 = sqrtf(fmaxf(bb, EPSV));
  const float sc2 = sinhf(n2) / n2;
  const float x2 = coshf(n2) * xd + sc2 * bd;
  const float x20u = coshf(n2) * x0 + sc2 * (coefb * (x0 + 1.0f));  // uniform x2[0]
  // xtan = logmap0(x)
  const float x0c = fmaxf(x20u, 1.0f + EPSV);
  const float coef = acoshf(x0c) * rsqrtf(x0c * x0c - 1.0f);
  x[(size_t)i * DD + d] = x2;
  xtan16[(size_t)i * DD + d] = (_Float16)((d == 0) ? 0.0f : coef * x2);
}

// ---------------- kernel 4: fused edge attention + scatter (persistent blocks) ----------------

__global__ void __launch_bounds__(256) k_edge(
    const _Float16* __restrict__ xtan16, const float* __restrict__ x,
    const float* __restrict__ ea, const int* __restrict__ row,
    const int* __restrict__ col, const float* __restrict__ emask,
    const _Float16* __restrict__ w1pp, const float* __restrict__ b1,
    const float* __restrict__ w2, const float* __restrict__ b2,
    float* __restrict__ agg, int nTiles, int E) {
  extern __shared__ char smem[];
  _Float16* Bt = (_Float16*)smem;                         // KT x 128 f16 fragment-major (73728 B)
  _Float16* At = (_Float16*)(smem + KT * DD * 2);         // 16 x KT f16  ( 9216 B)
  float* alphaS = (float*)(smem + KT * DD * 2 + 16 * KT * 2);
  float* accS   = alphaS + 16;
  float* attS   = accS + 16;

  const int tid = threadIdx.x;
  const int w = tid >> 5, ln = tid & 31;
  const int hi = ln >> 4, nn = ln & 15;
  const int nb = w * 16;

  // --- stage att_w1 (padded, fragment-major) into LDS once per block via async DMA ---
  for (int cb = tid; cb < (KT * DD) / 8; cb += 256) {
    const unsigned loff = (unsigned)(uintptr_t)(Bt + cb * 8);
    const unsigned long long gaddr = (unsigned long long)(uintptr_t)(w1pp + cb * 8);
    asm volatile("global_load_async_to_lds_b128 %0, %1, off"
                 :: "v"(loff), "v"(gaddr) : "memory");
  }
  asm volatile("s_wait_asynccnt 0x0" ::: "memory");
  __syncthreads();

  for (int t = blockIdx.x; t < nTiles; t += gridDim.x) {
    const int e0 = t * 16;
    if (t + gridDim.x < nTiles) {
      __builtin_prefetch(row + (size_t)(t + gridDim.x) * 16, 0, 0);
      __builtin_prefetch(col + (size_t)(t + gridDim.x) * 16, 0, 0);
    }
    // --- stage A tile: wave w handles edges 2w, 2w+1 ---
    #pragma unroll
    for (int s = 0; s < 2; ++s) {
      const int i = 2 * w + s;
      int e = e0 + i; if (e >= E) e = E - 1;
      const int r = row[e], ci = col[e];
      const unsigned* srcr = (const unsigned*)(xtan16 + (size_t)r * DD);
      const unsigned* srcc = (const unsigned*)(xtan16 + (size_t)ci * DD);
      unsigned* dst = (unsigned*)(At + i * KT);
      dst[ln] = srcr[ln];       dst[ln + 32] = srcr[ln + 32];
      dst[ln + 64] = srcc[ln];  dst[ln + 96] = srcc[ln + 32];
      // alpha = clip(-linner(xr,xc)) and geo = acosh(alpha)
      const float* xr = x + (size_t)r * DD;
      const float* xc = x + (size_t)ci * DD;
      const float4 xrv = *(const float4*)(xr + 4 * ln);
      const float4 xcv = *(const float4*)(xc + 4 * ln);
      float p = -(xrv.x * xcv.x + xrv.y * xcv.y + xrv.z * xcv.z + xrv.w * xcv.w);
      if (ln == 0) p += 2.0f * xrv.x * xcv.x;   // flip sign of the d==0 term
      float al = wave_sum32(p);
      al = fmaxf(al, 1.0f + EPSV);
      if (ln == 0) {
        alphaS[i] = al;
        At[i * KT + 256] = (_Float16)ea[(size_t)e * 2];
        At[i * KT + 257] = (_Float16)ea[(size_t)e * 2 + 1];
        At[i * KT + 258] = (_Float16)acoshf(al);
      }
      if (ln < 29) At[i * KT + 259 + ln] = (_Float16)0.0f;   // zero pad K 259..287
    }
    if (tid < 16) accS[tid] = 0.0f;
    __syncthreads();

    // --- hid = silu(att_in @ w1 + b1): each wave owns one 16x16 output tile ---
    v8f cacc = {};
    #pragma unroll
    for (int kk = 0; kk < KT / 32; ++kk) {
      const int k0 = kk * 32;
      v16h a;
      const _Float16* Ar = At + (ln & 15) * KT + k0;
      const v8h a0 = *(const v8h*)(Ar + 8 * hi);
      const v8h a1 = *(const v8h*)(Ar + 16 + 8 * hi);
      #pragma unroll
      for (int j = 0; j < 8; ++j) { a[j] = a0[j]; a[8 + j] = a1[j]; }
      const v16h b = *(const v16h*)(Bt + (size_t)(((kk * 8 + w) * 32 + ln) * 16));
      cacc = __builtin_amdgcn_wmma_f32_16x16x32_f16(false, a, false, b, (short)0, cacc,
                                                    false, false);
    }
    // --- epilogue: silu, dot with w2 (LDS f32 atomics reduce across 128 cols) ---
    const float b1c = b1[nb + nn];
    const float w2c = w2[nb + nn];
    #pragma unroll
    for (int r = 0; r < 8; ++r) {
      const float vv = cacc[r] + b1c;
      const float hval = vv / (1.0f + __expf(-vv));
      atomicAdd(&accS[r + 8 * hi], hval * w2c);
    }
    __syncthreads();
    if (tid < 16) {
      int e = e0 + tid; if (e >= E) e = E - 1;
      const float pre = accS[tid] + b2[0];
      attS[tid] = emask[e] / (1.0f + __expf(-pre));
    }
    __syncthreads();

    // --- loc = logmap(xr,xc); agg[row] += loc * att / 1000 ---
    #pragma unroll
    for (int s = 0; s < 2; ++s) {
      const int i = 2 * w + s;
      const int e = e0 + i;
      if (e < E) {
        const int r = row[e], ci = col[e];
        const float* xr = x + (size_t)r * DD;
        const float* xc = x + (size_t)ci * DD;
        const float al = alphaS[i];
        const float fac = acoshf(al) * rsqrtf(al * al - 1.0f) * attS[i] * (1.0f / NORMF);
        float* dstg = agg + (size_t)r * DD;
        #pragma unroll
        for (int j = 0; j < 4; ++j) {
          const int d = ln + 32 * j;
          atomicAdd(dstg + d, fac * (xc[d] - al * xr[d]));
        }
      }
    }
    __syncthreads();
  }
}

// ---------------- kernel 5: node epilogue (proju/expmap/layernorm/silu/expmap0) ----------------

__global__ void __launch_bounds__(128) k_node_post(const float* __restrict__ x,
                                                   const float* __restrict__ agg,
                                                   const float* __restrict__ g,
                                                   const float* __restrict__ be,
                                                   float* __restrict__ out) {
  __shared__ float red[128];
  __shared__ float sb;
  const int i = blockIdx.x, d = threadIdx.x;
  const float xd = x[(size_t)i * DD + d];
  const float ad = agg[(size_t)i * DD + d];
  // u = proju(x, agg)
  const float ip = blk_sum128((d == 0) ? -xd * ad : xd * ad, red);
  const float ud = ad + ip * xd;
  // x2 = expmap(x, u)
  const float uu = blk_sum128((d == 0) ? -ud * ud : ud * ud, red);
  const float n1 = sqrtf(fmaxf(uu, EPSV));
  const float x2 = coshf(n1) * xd + (sinhf(n1) / n1) * ud;
  if (d == 0) sb = x2;
  __syncthreads();
  // ht = logmap0(x2)
  const float x20 = fmaxf(sb, 1.0f + EPSV);
  const float coef1 = acoshf(x20) * rsqrtf(x20 * x20 - 1.0f);
  const float ht = (d == 0) ? 0.0f : coef1 * x2;
  // layernorm over ht[1:]
  const float m = blk_sum128(ht, red) * (1.0f / 127.0f);
  const float dv = (d == 0) ? 0.0f : (ht - m);
  const float var = blk_sum128(dv * dv, red) * (1.0f / 127.0f);
  const float sp = (d == 0) ? 0.0f : ((ht - m) * rsqrtf(var + 1e-5f) * g[d - 1] + be[d - 1]);
  // x3 = expmap0([0, sp])
  const float n2 = sqrtf(fmaxf(blk_sum128(sp * sp, red), EPSV));
  const float x3 = (d == 0) ? coshf(n2) : (sinhf(n2) / n2) * sp;
  // silu(logmap0(x3)), then expmap0
  const float x30 = fmaxf(coshf(n2), 1.0f + EPSV);
  const float coef2 = acoshf(x30) * rsqrtf(x30 * x30 - 1.0f);
  const float vv = (d == 0) ? 0.0f : coef2 * x3;
  const float sl = vv / (1.0f + __expf(-vv));
  const float n3 = sqrtf(fmaxf(blk_sum128((d == 0) ? 0.0f : sl * sl, red), EPSV));
  out[(size_t)i * DD + d] = (d == 0) ? coshf(n3) : (sinhf(n3) / n3) * sl;
}

// ---------------- host launcher ----------------

extern "C" void kernel_launch(void* const* d_in, const int* in_sizes, int n_in,
                              void* d_out, int out_size, void* d_ws, size_t ws_size,
                              hipStream_t stream) {
  (void)n_in; (void)out_size; (void)ws_size;
  const float* h     = (const float*)d_in[0];
  const float* ea    = (const float*)d_in[1];
  const int*   row   = (const int*)d_in[2];
  const int*   col   = (const int*)d_in[3];
  const float* emask = (const float*)d_in[5];
  const float* W     = (const float*)d_in[6];
  const float* bias  = (const float*)d_in[7];
  const float* w1    = (const float*)d_in[8];
  const float* b1    = (const float*)d_in[9];
  const float* w2    = (const float*)d_in[10];
  const float* b2    = (const float*)d_in[11];
  const float* lg    = (const float*)d_in[12];
  const float* lb    = (const float*)d_in[13];
  const int N = in_sizes[0] / DD;
  const int E = in_sizes[2];
  float* out = (float*)d_out;

  // workspace carve-out (~21 MB total)
  char* ws = (char*)d_ws;
  size_t off = 0;
  auto take = [&](size_t bytes) {
    char* p = ws + off;
    off = (off + bytes + 255) & ~(size_t)255;
    return p;
  };
  _Float16* u16    = (_Float16*)take((size_t)N * DD * 2);
  float*    xt     = (float*)take((size_t)N * DD * 4);
  float*    x      = (float*)take((size_t)N * DD * 4);
  _Float16* xtan16 = (_Float16*)take((size_t)N * DD * 2);
  float*    agg    = (float*)take((size_t)N * DD * 4);
  _Float16* W16p   = (_Float16*)take((size_t)DD * DD * 2);
  _Float16* w1pp   = (_Float16*)take((size_t)KT * DD * 2);

  const int total = N * DD;
  k_init<<<(total + 255) / 256, 256, 0, stream>>>(W, w1, W16p, w1pp, agg, total);
  k_node_pre<<<(total + 255) / 256, 256, 0, stream>>>(h, u16, total);
  k_gemm_xw<<<(N + 15) / 16, 256, 0, stream>>>(u16, W16p, xt, N);
  k_node_mid<<<N, 128, 0, stream>>>(xt, bias, x, xtan16);
  const int nTiles = (E + 15) / 16;
  const size_t smem = (size_t)KT * DD * 2 + (size_t)16 * KT * 2 + 3 * 16 * 4;
  k_edge<<<640, 256, smem, stream>>>(xtan16, x, ea, row, col, emask, w1pp, b1, w2, b2,
                                     agg, nTiles, E);
  k_node_post<<<N, 128, 0, stream>>>(x, agg, lg, lb, out);
}
